// DecoderTreeRNN_42013370089581
// MI455X (gfx1250) — matrix operand order repaired and checked by reading, not compiled, not run on hardware
//
#include <hip/hip_runtime.h>
#include <hip/hip_bf16.h>
#include <stdint.h>

// ---------------------------------------------------------------------------
// DecoderTreeRNN for MI455X (gfx1250): bf16 WMMA, f32 accumulate, f32 GRU
// state between levels, software-pipelined fragment loads, TDM-staged B tile
// for the output GEMM (tensor_load_to_lds + s_wait_tensorcnt + ds_load),
// fused gates, single-pass online log-sum-exp.
// ---------------------------------------------------------------------------

typedef __attribute__((ext_vector_type(16))) __bf16        v16bf;
typedef __attribute__((ext_vector_type(8)))  float         v8f;
typedef __attribute__((ext_vector_type(4)))  unsigned int  u32x4;
typedef __attribute__((ext_vector_type(8)))  int           i32x8;
typedef __attribute__((ext_vector_type(4)))  int           i32x4;

struct alignas(16) U128 { unsigned int x, y, z, w; };
union FragBF { v16bf v; U128 q[2]; };

static __device__ __forceinline__ v16bf ldfrag(const unsigned short* p0,
                                               const unsigned short* p1) {
  FragBF f;
  f.q[0] = *reinterpret_cast<const U128*>(p0);
  f.q[1] = *reinterpret_cast<const U128*>(p1);
  return f.v;
}

static __device__ __forceinline__ v8f wmma_bf16(v16bf a, v16bf b, v8f c) {
  // D = A(16x32 bf16) x B(32x16 bf16) + C(16x16 f32)
  return __builtin_amdgcn_wmma_f32_16x16x32_bf16(
      /*neg_a=*/false, a, /*neg_b=*/false, b,
      /*c_mod=*/(short)0, c, /*reuse_a=*/false, /*reuse_b=*/false);
}

static __device__ __forceinline__ unsigned short f2bf(float f) {
  unsigned u = __float_as_uint(f);
  u += 0x7FFFu + ((u >> 16) & 1u);   // round-to-nearest-even
  return (unsigned short)(u >> 16);
}

static __device__ __forceinline__ float sigmoidf_fast(float x) {
  return 1.0f / (1.0f + __expf(-x));
}

// Problem constants (fixed by the reference file).
constexpr int  Bsz   = 256;
constexpr int  Hdim  = 512;
constexpr int  Vdim  = 10000;
constexpr int  DEPTH = 6;
constexpr int  Mfin  = Bsz * (1 << DEPTH);       // 16384 final rows

// ---------------------------------------------------------------------------
// f32 -> bf16 weight conversion (grid-stride)
// ---------------------------------------------------------------------------
__global__ void conv_bf16_kernel(const float* __restrict__ src,
                                 unsigned short* __restrict__ dst, int n) {
  for (int i = blockIdx.x * blockDim.x + threadIdx.x; i < n;
       i += gridDim.x * blockDim.x)
    dst[i] = f2bf(src[i]);
}

// ---------------------------------------------------------------------------
// h0 = relu(encoding), stored f32 + bf16 mirror
// ---------------------------------------------------------------------------
__global__ void relu_init_kernel(const float* __restrict__ enc,
                                 float* __restrict__ hf,
                                 unsigned short* __restrict__ hb, int n) {
  int i = blockIdx.x * blockDim.x + threadIdx.x;
  if (i < n) {
    float v = fmaxf(enc[i], 0.0f);
    hf[i] = v;
    hb[i] = f2bf(v);
  }
}

// ---------------------------------------------------------------------------
// One GRU tree level: rows Min parents -> 2*Min children (interleaved L,R).
// Block tile: 64 rows x 32 cols; 8 waves as 4(row) x 2(col); each wave owns a
// 16x16 C tile with 6 accumulators (L/R x {r,z,n}).  gh = h @ W^T + b_hh.
// Fragment loads for k+1 issue before the 6 WMMAs of k (double buffering).
// ---------------------------------------------------------------------------
__global__ __launch_bounds__(256) void gru_level_kernel(
    const unsigned short* __restrict__ hbf,   // (Min,512) bf16
    const float*          __restrict__ hf32,  // (Min,512) f32
    const unsigned short* __restrict__ WLbf,  // (1536,512) bf16
    const unsigned short* __restrict__ WRbf,  // (1536,512) bf16
    const float* __restrict__ bihL, const float* __restrict__ bhhL,
    const float* __restrict__ bihR, const float* __restrict__ bhhR,
    float*          __restrict__ outF,        // (2*Min,512) f32
    unsigned short* __restrict__ outBF)       // (2*Min,512) bf16
{
  const int lane = threadIdx.x & 31;
  const int wave = threadIdx.x >> 5;
  const int wy = wave >> 1, wx = wave & 1;
  const int half = lane >> 4, l16 = lane & 15;

  const int m0 = blockIdx.y * 64 + wy * 16;       // row tile origin
  const int n0 = blockIdx.x * 32 + wx * 16;       // col tile origin
  const int arow = m0 + l16;                      // A-fragment row (M = lane%16)
  const int bcol = n0 + l16;                      // B-fragment col (N = lane%16)

  const unsigned short* pa  = hbf  + (size_t)arow * Hdim + half * 16;
  const unsigned short* pbL = WLbf + (size_t)bcol * Hdim + half * 16;
  const unsigned short* pbR = WRbf + (size_t)bcol * Hdim + half * 16;
  const size_t G = (size_t)Hdim * Hdim;           // 512 rows per gate block

  v8f aLr = {}, aLz = {}, aLn = {}, aRr = {}, aRz = {}, aRn = {};

  // prologue: fragments for k0 = 0
  v16bf a  = ldfrag(pa, pa + 8);
  v16bf b0 = ldfrag(pbL,         pbL + 8);
  v16bf b1 = ldfrag(pbL + G,     pbL + G + 8);
  v16bf b2 = ldfrag(pbL + 2 * G, pbL + 2 * G + 8);
  v16bf b3 = ldfrag(pbR,         pbR + 8);
  v16bf b4 = ldfrag(pbR + G,     pbR + G + 8);
  v16bf b5 = ldfrag(pbR + 2 * G, pbR + 2 * G + 8);

#pragma unroll
  for (int k0 = 0; k0 < Hdim; k0 += 32) {
    v16bf an, c0, c1, c2, c3, c4, c5;
    if (k0 + 32 < Hdim) {                         // issue next-iter loads first
      const int kn = k0 + 32;
      an = ldfrag(pa + kn, pa + kn + 8);
      c0 = ldfrag(pbL + kn,         pbL + kn + 8);
      c1 = ldfrag(pbL + G + kn,     pbL + G + kn + 8);
      c2 = ldfrag(pbL + 2 * G + kn, pbL + 2 * G + kn + 8);
      c3 = ldfrag(pbR + kn,         pbR + kn + 8);
      c4 = ldfrag(pbR + G + kn,     pbR + G + kn + 8);
      c5 = ldfrag(pbR + 2 * G + kn, pbR + 2 * G + kn + 8);
    }
    aLr = wmma_bf16(a, b0, aLr);
    aLz = wmma_bf16(a, b1, aLz);
    aLn = wmma_bf16(a, b2, aLn);
    aRr = wmma_bf16(a, b3, aRr);
    aRz = wmma_bf16(a, b4, aRz);
    aRn = wmma_bf16(a, b5, aRn);
    a = an; b0 = c0; b1 = c1; b2 = c2; b3 = c3; b4 = c4; b5 = c5;
  }

  // Per-lane column is fixed (= bcol): load the 12 biases once.
  const float bLr  = bihL[bcol]            + bhhL[bcol];
  const float bLz  = bihL[Hdim + bcol]     + bhhL[Hdim + bcol];
  const float bLni = bihL[2 * Hdim + bcol];
  const float bLnh = bhhL[2 * Hdim + bcol];
  const float bRr  = bihR[bcol]            + bhhR[bcol];
  const float bRz  = bihR[Hdim + bcol]     + bhhR[Hdim + bcol];
  const float bRni = bihR[2 * Hdim + bcol];
  const float bRnh = bhhR[2 * Hdim + bcol];

#pragma unroll
  for (int v = 0; v < 8; ++v) {
    const int mrow = m0 + half * 8 + v;            // C layout: M = v + 8*(lane/16)
    const float hv = hf32[(size_t)mrow * Hdim + bcol];
    const size_t orow = (size_t)2 * mrow;          // children interleave: 2m, 2m+1

    // left child
    float r  = sigmoidf_fast(bLr + aLr[v]);
    float z  = sigmoidf_fast(bLz + aLz[v]);
    float nn = tanhf(bLni + r * (aLn[v] + bLnh));
    float o  = (1.0f - z) * nn + z * hv;
    outF [orow * Hdim + bcol] = o;
    outBF[orow * Hdim + bcol] = f2bf(o);

    // right child
    r  = sigmoidf_fast(bRr + aRr[v]);
    z  = sigmoidf_fast(bRz + aRz[v]);
    nn = tanhf(bRni + r * (aRn[v] + bRnh));
    o  = (1.0f - z) * nn + z * hv;
    outF [(orow + 1) * Hdim + bcol] = o;
    outBF[(orow + 1) * Hdim + bcol] = f2bf(o);
  }
}

// ---------------------------------------------------------------------------
// logits(16384 x 10000) = H(16384x512) x Wout^T + b_out.
// The block's B operand (16 W_out rows x K=512 = 16 KB) is staged into LDS
// once via the Tensor Data Mover (one 2D D#, issued by wave 0, waited with
// s_wait_tensorcnt, released by the workgroup barrier), then all 8 waves read
// B fragments with ds_load_b128 across the K loop.  Each wave computes a
// 64x16 strip: 4 A fragments (global, double-buffered) share one LDS B
// fragment per K-step -> 4 WMMA per 4 global-load pairs.
// ---------------------------------------------------------------------------
__global__ __launch_bounds__(256) void out_gemm_kernel(
    const unsigned short* __restrict__ hbf,   // (16384,512) bf16
    const unsigned short* __restrict__ wobf,  // (10000,512) bf16
    const float* __restrict__ bout,           // (10000,)
    float* __restrict__ out)                  // (16384,10000)
{
  __shared__ unsigned short ldsB[16 * Hdim];       // 16 KB: 16 cols x K=512

  const int tid  = threadIdx.x;
  const int lane = tid & 31;
  const int wave = tid >> 5;
  const int half = lane >> 4, l16 = lane & 15;

  const int m0 = (blockIdx.y * 8 + wave) * 64;     // 64-row strip per wave
  const int n0 = blockIdx.x * 16;
  const int bcol = n0 + l16;

  // ---- Stage W_out rows [n0, n0+16) x 512 into LDS --------------------------
#if __has_builtin(__builtin_amdgcn_tensor_load_to_lds) && \
    __has_builtin(__builtin_amdgcn_s_wait_tensorcnt)
  if (wave == 0) {
    const unsigned long long gaddr =
        (unsigned long long)(uintptr_t)(wobf + (size_t)n0 * Hdim);
    const unsigned ldsoff =
        (unsigned)(uintptr_t)(__attribute__((address_space(3))) unsigned short*)ldsB;
    u32x4 g0;
    g0[0] = 1u;                                    // count=1, user descriptor
    g0[1] = ldsoff;                                // lds_addr (bytes)
    g0[2] = (unsigned)gaddr;                       // global_addr[31:0]
    g0[3] = (unsigned)((gaddr >> 32) & 0x01FFFFFFull)  // global_addr[56:32]
            | (2u << 30);                          // type = 2 ("image")
    i32x8 g1;
    g1[0] = (int)(1u << 16);                       // wg_mask=0, data_size=1 (2B)
    g1[1] = (int)(512u << 16);                     // tensor_dim0 = 512 (lo 16)
    g1[2] = (int)(16u  << 16);                     // t_dim0 hi=0 | tensor_dim1=16
    g1[3] = (int)(512u << 16);                     // t_dim1 hi=0 | tile_dim0=512
    g1[4] = 16;                                    // tile_dim1=16, tile_dim2=0
    g1[5] = 512;                                   // tensor_dim0_stride[31:0]
    g1[6] = (int)(512u << 16);                     // t0s hi=0 | t1_stride lo=512
    g1[7] = 0;                                     // t1_stride[47:16]=0
    i32x4 z4 = {0, 0, 0, 0};                       // <=2D: groups 2/3 unused
    i32x8 z8 = {0, 0, 0, 0, 0, 0, 0, 0};           // 6-arg form extra group
    __builtin_amdgcn_tensor_load_to_lds(g0, g1, z4, z4, z8, 0);
    __builtin_amdgcn_s_wait_tensorcnt(0);
  }
  __syncthreads();
#else
  // Fallback: cooperative VMEM->LDS copy (b128 granules).
  for (int i = tid * 8; i < 16 * Hdim; i += 256 * 8)
    *reinterpret_cast<U128*>(ldsB + i) =
        *reinterpret_cast<const U128*>(wobf + (size_t)n0 * Hdim + i);
  __syncthreads();
#endif

  const unsigned short* pa0 = hbf + (size_t)(m0 +  0 + l16) * Hdim + half * 8;
  const unsigned short* pa1 = hbf + (size_t)(m0 + 16 + l16) * Hdim + half * 8;
  const unsigned short* pa2 = hbf + (size_t)(m0 + 32 + l16) * Hdim + half * 8;
  const unsigned short* pa3 = hbf + (size_t)(m0 + 48 + l16) * Hdim + half * 8;
  const unsigned short* pb  = ldsB + l16 * Hdim + half * 8;   // LDS B rows

  v8f c0 = {}, c1 = {}, c2 = {}, c3 = {};

  v16bf a0 = ldfrag(pa0, pa0 + 16);
  v16bf a1 = ldfrag(pa1, pa1 + 16);
  v16bf a2 = ldfrag(pa2, pa2 + 16);
  v16bf a3 = ldfrag(pa3, pa3 + 16);

#pragma unroll
  for (int k0 = 0; k0 < Hdim; k0 += 32) {
    v16bf b = ldfrag(pb + k0, pb + k0 + 16);        // ds_load_b128 x2
    v16bf n0f, n1f, n2f, n3f;
    if (k0 + 32 < Hdim) {                          // next-iter A (global)
      const int kn = k0 + 32;
      n0f = ldfrag(pa0 + kn, pa0 + kn + 16);
      n1f = ldfrag(pa1 + kn, pa1 + kn + 16);
      n2f = ldfrag(pa2 + kn, pa2 + kn + 16);
      n3f = ldfrag(pa3 + kn, pa3 + kn + 16);
    }
    c0 = wmma_bf16(a0, b, c0);
    c1 = wmma_bf16(a1, b, c1);
    c2 = wmma_bf16(a2, b, c2);
    c3 = wmma_bf16(a3, b, c3);
    a0 = n0f; a1 = n1f; a2 = n2f; a3 = n3f;
  }

  const float bo = bout[bcol];
#pragma unroll
  for (int v = 0; v < 8; ++v) {
    const int mr = half * 8 + v;
    out[(size_t)(m0 +  0 + mr) * Vdim + bcol] = c0[v] + bo;
    out[(size_t)(m0 + 16 + mr) * Vdim + bcol] = c1[v] + bo;
    out[(size_t)(m0 + 32 + mr) * Vdim + bcol] = c2[v] + bo;
    out[(size_t)(m0 + 48 + mr) * Vdim + bcol] = c3[v] + bo;
  }
}

// ---------------------------------------------------------------------------
// Per-row online log-sum-exp (single read pass): stats[m] = max + log(sumexp)
// ---------------------------------------------------------------------------
__global__ __launch_bounds__(256) void lse_kernel(const float* __restrict__ logits,
                                                  float* __restrict__ stats) {
  const int m = blockIdx.x;
  const int tid = threadIdx.x;
  const float* row = logits + (size_t)m * Vdim;

  float mx = -INFINITY, s = 0.0f;
  for (int v = tid; v < Vdim; v += 256) {
    float x = row[v];
    float nm = fmaxf(mx, x);
    s = s * __expf(mx - nm) + __expf(x - nm);
    mx = nm;
  }

  __shared__ float sm[256];
  __shared__ float ss[256];
  sm[tid] = mx; ss[tid] = s;
  __syncthreads();
  for (int o = 128; o > 0; o >>= 1) {
    if (tid < o) {
      float m2 = sm[tid + o], s2 = ss[tid + o];
      float nm = fmaxf(sm[tid], m2);
      ss[tid] = ss[tid] * __expf(sm[tid] - nm) + s2 * __expf(m2 - nm);
      sm[tid] = nm;
    }
    __syncthreads();
  }
  if (tid == 0) stats[m] = sm[0] + __logf(ss[0]);
}

// ---------------------------------------------------------------------------
// Normalize: out[m,v] -= stats[m]  (float4; 10000 % 4 == 0 so no row splits)
// ---------------------------------------------------------------------------
struct alignas(16) F4 { float x, y, z, w; };

__global__ __launch_bounds__(256) void lsm_kernel(float* __restrict__ logits,
                                                  const float* __restrict__ stats) {
  const size_t i = (size_t)blockIdx.x * 256 + threadIdx.x;   // float4 index
  const size_t n4 = (size_t)Mfin * Vdim / 4;
  if (i >= n4) return;
  const size_t e = i * 4;
  const float sh = stats[e / Vdim];
  F4* p = reinterpret_cast<F4*>(logits) + i;
  F4 v = *p;
  v.x -= sh; v.y -= sh; v.z -= sh; v.w -= sh;
  *p = v;
}

// ---------------------------------------------------------------------------
// Host launcher
// ---------------------------------------------------------------------------
extern "C" void kernel_launch(void* const* d_in, const int* in_sizes, int n_in,
                              void* d_out, int out_size, void* d_ws, size_t ws_size,
                              hipStream_t stream) {
  (void)in_sizes; (void)n_in; (void)out_size; (void)ws_size;

  const float* enc  = (const float*)d_in[0];
  const float* WL   = (const float*)d_in[1];
  const float* bihL = (const float*)d_in[2];
  const float* bhhL = (const float*)d_in[3];
  const float* WR   = (const float*)d_in[4];
  const float* bihR = (const float*)d_in[5];
  const float* bhhR = (const float*)d_in[6];
  const float* WO   = (const float*)d_in[7];
  const float* bout = (const float*)d_in[8];
  // d_in[9] = depth (device scalar); fixed to 6 by the reference file.

  char* ws = (char*)d_ws;
  constexpr size_t SZ_WHH  = (size_t)3 * Hdim * Hdim * 2;     // 1,572,864
  constexpr size_t SZ_WO   = (size_t)Vdim * Hdim * 2;         // 10,240,000
  constexpr size_t SZ_HF   = (size_t)Mfin * Hdim * 4;         // 33,554,432
  constexpr size_t SZ_HB   = (size_t)Mfin * Hdim * 2;         // 16,777,216
  size_t off = 0;
  unsigned short* WLbf = (unsigned short*)(ws + off); off += SZ_WHH;
  unsigned short* WRbf = (unsigned short*)(ws + off); off += SZ_WHH;
  unsigned short* WObf = (unsigned short*)(ws + off); off += SZ_WO;
  float*          hAf  = (float*)(ws + off);          off += SZ_HF;
  float*          hBf  = (float*)(ws + off);          off += SZ_HF;
  unsigned short* hAb  = (unsigned short*)(ws + off); off += SZ_HB;
  unsigned short* hBb  = (unsigned short*)(ws + off); off += SZ_HB;
  float*          stats = (float*)(ws + off);

  // Weight precision conversion (L2-resident thereafter).
  {
    const int nW = 3 * Hdim * Hdim;
    conv_bf16_kernel<<<1024, 256, 0, stream>>>(WL, WLbf, nW);
    conv_bf16_kernel<<<1024, 256, 0, stream>>>(WR, WRbf, nW);
    conv_bf16_kernel<<<4096, 256, 0, stream>>>(WO, WObf, Vdim * Hdim);
  }

  // h0 = relu(encoding)
  relu_init_kernel<<<(Bsz * Hdim + 255) / 256, 256, 0, stream>>>(
      enc, hAf, hAb, Bsz * Hdim);

  // 6 tree levels (ping-pong buffers); 6 swaps -> final state back in A.
  float* curF = hAf; float* nxtF = hBf;
  unsigned short* curB = hAb; unsigned short* nxtB = hBb;
  for (int d = 0; d < DEPTH; ++d) {
    const int Min = Bsz << d;                      // parent rows: 256..8192
    dim3 grid(Hdim / 32, Min / 64);
    gru_level_kernel<<<grid, 256, 0, stream>>>(
        curB, curF, WLbf, WRbf, bihL, bhhL, bihR, bhhR, nxtF, nxtB);
    float* tf = curF; curF = nxtF; nxtF = tf;
    unsigned short* tb = curB; curB = nxtB; nxtB = tb;
  }

  // Output projection + bias into d_out.
  {
    dim3 grid(Vdim / 16, Mfin / 512);              // 625 x 32
    out_gemm_kernel<<<grid, 256, 0, stream>>>(curB, WObf, bout, (float*)d_out);
  }

  // log_softmax: one fused max/sum pass, then normalize.
  lse_kernel<<<Mfin, 256, 0, stream>>>((const float*)d_out, stats);
  {
    const size_t n4 = (size_t)Mfin * Vdim / 4;     // 40,960,000
    lsm_kernel<<<(unsigned)((n4 + 255) / 256), 256, 0, stream>>>(
        (float*)d_out, stats);
  }
}
